// CrissCrossAttention_12386685682050
// MI455X (gfx1250) — matrix operand order repaired and checked by reading, not compiled, not run on hardware
//
#include <hip/hip_runtime.h>
#include <hip/hip_bf16.h>

// ---------------- CDNA5 (gfx1250) criss-cross attention ----------------
// wave32, WMMA bf16 16x16x32, f32 accumulate.
// LDS tile layouts: A tiles [M][K], B tiles [N][K]  -> all fragment loads are
// contiguous 2x ds_load_b128 per lane.

typedef __bf16 bf16_t;
typedef __attribute__((ext_vector_type(16))) __bf16 v16bf;
typedef __attribute__((ext_vector_type(8)))  float  v8f;

constexpr int B_    = 4;
constexpr int C_    = 512;
constexpr int H_    = 128;
constexpr int W_    = 128;
constexpr int HW_   = H_ * W_;
constexpr int MQKV_ = 640;   // 64 q + 64 k + 512 v rows

#define WMMA_BF16(A, Bm, Cacc) \
  __builtin_amdgcn_wmma_f32_16x16x32_bf16(false, (A), false, (Bm), (short)0, (Cacc), false, false)

// ---- async global->LDS copy (signature per hipcc diagnostic: v4i pointers) ----
#if defined(__has_builtin)
#if __has_builtin(__builtin_amdgcn_global_load_async_to_lds_b128) && \
    __has_builtin(__builtin_amdgcn_s_wait_asynccnt)
#define USE_ASYNC_LDS 1
#endif
#if __has_builtin(__builtin_amdgcn_tensor_load_to_lds) && \
    __has_builtin(__builtin_amdgcn_s_wait_tensorcnt)
#define USE_TDM 1
#endif
#endif

#if defined(USE_ASYNC_LDS)
typedef int async_v4i __attribute__((vector_size(16)));
typedef __attribute__((address_space(1))) async_v4i* gbl_v4i_p;
typedef __attribute__((address_space(3))) async_v4i* lds_v4i_p;
__device__ __forceinline__ void ldsCopy16(void* lds, const void* g) {
  gbl_v4i_p gp = (gbl_v4i_p)(unsigned long long)(uintptr_t)g;
  lds_v4i_p lp = (lds_v4i_p)(unsigned)(uintptr_t)lds;
  __builtin_amdgcn_global_load_async_to_lds_b128(gp, lp, 0, 0);
}
__device__ __forceinline__ void ldsCopyFence() { __builtin_amdgcn_s_wait_asynccnt(0); }
#else
__device__ __forceinline__ void ldsCopy16(void* lds, const void* g) {
  *(uint4*)lds = *(const uint4*)g;
}
__device__ __forceinline__ void ldsCopyFence() {}
#endif

#if defined(USE_TDM)
// Tensor Data Mover: 2D bf16 tile (rows x rowElems), row stride in elements,
// LDS destination padded 16B every 256B (matches [128][136] bf16 layout).
// 6-arg builtin form: (u32x4 g0, i32x8 g1, i32x4 g2, i32x4 g3, i32x8, i32 cpol)
typedef unsigned int tdm_u32x4 __attribute__((vector_size(16)));
typedef int          tdm_i32x8 __attribute__((vector_size(32)));
typedef int          tdm_i32x4 __attribute__((vector_size(16)));
__device__ __forceinline__ void tdm_load_2d_bf16(void* lds, const void* gsrc,
                                                 unsigned rowElems, unsigned rows,
                                                 unsigned rowStrideElems) {
  const unsigned long long ga = (unsigned long long)(uintptr_t)gsrc;
  const unsigned la = (unsigned)(uintptr_t)lds;
  tdm_u32x4 g0;
  g0[0] = 1u;                                         // count=1 (valid user D#)
  g0[1] = la;                                         // lds_addr (bytes)
  g0[2] = (unsigned)(ga & 0xFFFFFFFFu);               // global_addr[31:0]
  g0[3] = (unsigned)((ga >> 32) & 0x01FFFFFFu)        // global_addr[56:32]
        | (2u << 30);                                 // type=2 ("image")
  tdm_i32x8 g1;
  g1[0] = (int)((1u << 16)                            // data_size=1 -> 2 bytes
        | (1u << 20)                                  // pad_enable
        | (5u << 22)                                  // pad_interval: 64 DWORDs (256B)
        | (3u << 25));                                // pad_amount: 4 DWORDs (16B)
  g1[1] = (int)((rowStrideElems & 0xFFFFu) << 16);    // tensor_dim0[15:0]
  g1[2] = (int)(((rowStrideElems >> 16) & 0xFFFFu)    // tensor_dim0[31:16]
        | ((rows & 0xFFFFu) << 16));                  // tensor_dim1[15:0]
  g1[3] = (int)(((rows >> 16) & 0xFFFFu)              // tensor_dim1[31:16]
        | ((rowElems & 0xFFFFu) << 16));              // tile_dim0
  g1[4] = (int)(rows & 0xFFFFu);                      // tile_dim1 (tile_dim2=0)
  g1[5] = (int)rowStrideElems;                        // tensor_dim0_stride[31:0]
  g1[6] = 0;                                          // stride hi / dim1_stride
  g1[7] = 0;
  tdm_i32x4 z4 = {0, 0, 0, 0};                        // 2D tensor: groups 2/3 unused
  tdm_i32x8 z8 = {0, 0, 0, 0, 0, 0, 0, 0};
  __builtin_amdgcn_tensor_load_to_lds(g0, g1, z4, z4, z8, 0);
}
#endif

// ---- fragment loaders (contiguous k order, not the ISA table order; A and B match) ----
// A (16x32) from LDS [M][K]: lanes 0-15 row M=lane, kbase 0; lanes 16-31 row
// M=lane-16, kbase 16. element e -> K = kbase + e  (one contiguous 32B run)
__device__ __forceinline__ v16bf load_fragA(const bf16_t* p, int row0, int k0, int ld) {
  const int lane = threadIdx.x & 31;
  const int r  = row0 + (lane & 15);
  const int kb = k0 + ((lane & 16) ? 16 : 0);
  v16bf a;
#pragma unroll
  for (int e = 0; e < 16; ++e) {
    const int kk = kb + e;
    a[e] = p[(size_t)r * ld + kk];
  }
  return a;
}
// B (32x16) from LDS [N][K]: lanes 0-15 col N=lane, K=k0..k0+15; lanes 16-31
// col N=lane-16, K=k0+16..k0+31.  (one contiguous 32B run)
__device__ __forceinline__ v16bf load_fragB(const bf16_t* p, int k0, int col0, int ld) {
  const int lane = threadIdx.x & 31;
  const int c  = col0 + (lane & 15);
  const int kb = k0 + ((lane & 16) ? 16 : 0);
  v16bf b;
#pragma unroll
  for (int e = 0; e < 16; ++e) b[e] = p[(size_t)c * ld + kb + e];
  return b;
}
// C/D (16x16 f32): lanes 0-15: N=lane, VGPR r -> M=r; lanes 16-31: M=r+8.

// =============== K1: fused QKV projection ===============
// qkv[b][m][p] = sum_c Wsel[m][c] * x[b][c][p]   (m<64:Wq, <128:Wk, else Wv)
__global__ __launch_bounds__(256) void qkv_proj_kernel(
    const float* __restrict__ x, const float* __restrict__ Wq,
    const float* __restrict__ Wk, const float* __restrict__ Wv,
    bf16_t* __restrict__ qkv)
{
  __shared__ __align__(16) bf16_t As[64][32 + 8];    // [m][k]
  __shared__ __align__(16) bf16_t Bs[128][32 + 8];   // [n][k]
  const int b  = blockIdx.z;
  const int m0 = blockIdx.y * 64;
  const int n0 = blockIdx.x * 128;
  const int t  = threadIdx.x;
  const int wave = t >> 5;
  const int wm = (wave >> 2) * 32;     // 2 waves along M
  const int wn = (wave & 3) * 32;      // 4 waves along N
  const float* xb = x + (size_t)b * C_ * HW_;
  v8f acc[2][2] = {};

  // staging thread mappings (constant across k-steps)
  const int ar  = t >> 2;              // A: row, 8 consecutive k
  const int ac  = (t & 3) * 8;
  const int bn  = t & 127;             // B: col n, 16 consecutive k
  const int bk  = (t >> 7) * 16;

  for (int k0 = 0; k0 < C_; k0 += 32) {
    __syncthreads();
    { // A: 64x32 weights -> As[m][k]
      const int mg = m0 + ar;
      const float* wrow = (mg < 64)  ? (Wq + (size_t)mg * C_)
                        : (mg < 128) ? (Wk + (size_t)(mg - 64) * C_)
                                     : (Wv + (size_t)(mg - 128) * C_);
#pragma unroll
      for (int u = 0; u < 8; ++u) As[ar][ac + u] = (bf16_t)wrow[k0 + ac + u];
    }
    { // B: x tile -> Bs[n][k]; lane-coalesced reads (n contiguous across lanes)
      const float* xcol = xb + (size_t)k0 * HW_ + n0 + bn;
#pragma unroll
      for (int u = 0; u < 16; ++u)
        Bs[bn][bk + u] = (bf16_t)xcol[(size_t)(bk + u) * HW_];
      if (k0 + 32 < C_)
        __builtin_prefetch(xcol + (size_t)(bk + 32) * HW_, 0, 3);
    }
    __syncthreads();
#pragma unroll
    for (int mi = 0; mi < 2; ++mi) {
      v16bf a = load_fragA(&As[0][0], wm + mi * 16, 0, 40);
#pragma unroll
      for (int ni = 0; ni < 2; ++ni) {
        v16bf bb = load_fragB(&Bs[0][0], 0, wn + ni * 16, 40);
        acc[mi][ni] = WMMA_BF16(a, bb, acc[mi][ni]);
      }
    }
  }
  const int lane = t & 31;
  const int ncol = lane & 15;
  const int moff = (lane & 16) ? 8 : 0;
#pragma unroll
  for (int mi = 0; mi < 2; ++mi)
#pragma unroll
    for (int ni = 0; ni < 2; ++ni)
#pragma unroll
      for (int r = 0; r < 8; ++r) {
        const int m = m0 + wm + mi * 16 + moff + r;
        const int n = n0 + wn + ni * 16 + ncol;
        qkv[((size_t)b * MQKV_ + m) * HW_ + n] = (bf16_t)acc[mi][ni][r];
      }
}

// =============== K2: energies E = Q^T K (per column / per row) ===============
// mode 0: fixed w=pos, i,j run over H (diag masked); mode 1: fixed h=pos.
__global__ __launch_bounds__(256) void energy_kernel(
    const bf16_t* __restrict__ qkv, float* __restrict__ eH, float* __restrict__ eW)
{
  __shared__ __align__(16) bf16_t Qs[128][64 + 8];    // [i][m]  A = Q^T
  __shared__ __align__(16) bf16_t Ks[128][64 + 8];    // [j][m]  B as [N][K]
  const int pos  = blockIdx.x;
  const int mode = blockIdx.y;
  const int b    = blockIdx.z;
  const int t    = threadIdx.x;
  const size_t qbase = (size_t)b * MQKV_ * HW_;
  const size_t kbase = qbase + (size_t)64 * HW_;
  const int stride = mode ? 1 : W_;
  const int off    = mode ? pos * W_ : pos;

  // stage: per thread fixed i, 8 consecutive m; lane-coalesced in i
  for (int idx = t; idx < 1024; idx += 256) {
    const int i  = idx & 127;
    const int mc = (idx >> 7) * 8;
#pragma unroll
    for (int u = 0; u < 8; ++u) {
      const size_t src = (size_t)(mc + u) * HW_ + off + (size_t)i * stride;
      Qs[i][mc + u] = qkv[qbase + src];
      Ks[i][mc + u] = qkv[kbase + src];
    }
  }
  __syncthreads();

  const int wave = t >> 5;
  const int wi = (wave >> 2) * 64;     // 2 waves along i
  const int wj = (wave & 3) * 32;      // 4 waves along j
  v8f acc[4][2] = {};
#pragma unroll
  for (int k0 = 0; k0 < 64; k0 += 32)
#pragma unroll
    for (int mi = 0; mi < 4; ++mi) {
      v16bf a = load_fragA(&Qs[0][0], wi + mi * 16, k0, 72);
#pragma unroll
      for (int ni = 0; ni < 2; ++ni) {
        v16bf bb = load_fragB(&Ks[0][0], k0, wj + ni * 16, 72);
        acc[mi][ni] = WMMA_BF16(a, bb, acc[mi][ni]);
      }
    }

  float* E = (mode ? eW : eH) + ((size_t)b * 128 + pos) * 128 * 128;
  const int lane = t & 31;
  const int jcol = lane & 15;
  const int moff = (lane & 16) ? 8 : 0;
#pragma unroll
  for (int mi = 0; mi < 4; ++mi)
#pragma unroll
    for (int ni = 0; ni < 2; ++ni)
#pragma unroll
      for (int r = 0; r < 8; ++r) {
        const int i = wi + mi * 16 + moff + r;
        const int j = wj + ni * 16 + jcol;
        float v = acc[mi][ni][r];
        if (mode == 0 && i == j) v = -1e30f;
        E[(size_t)i * 128 + j] = v;
      }
}

// =============== K3: joint softmax over [eH row | eW row] (256 vals) ===============
// One wave per pixel; writes bf16 attention in place over the f32 rows
// (rows exclusively owned by this wave; bf16 row stride = 256 elements).
__global__ __launch_bounds__(256) void softmax_kernel(
    float* __restrict__ eH, float* __restrict__ eW)
{
  const int wgl  = blockIdx.x * 8 + (threadIdx.x >> 5);
  const int lane = threadIdx.x & 31;
  const int b   = wgl >> 14;
  const int pix = wgl & 16383;
  const int h = pix >> 7;
  const int w = pix & 127;
  float* rH = eH + (((size_t)b * 128 + w) * 128 + h) * 128;  // eH[b][w][h][:]
  float* rW = eW + (((size_t)b * 128 + h) * 128 + w) * 128;  // eW[b][h][w][:]

  float vh[4], vw[4];
#pragma unroll
  for (int u = 0; u < 4; ++u) { vh[u] = rH[lane * 4 + u]; vw[u] = rW[lane * 4 + u]; }
  float mx = -1e30f;
#pragma unroll
  for (int u = 0; u < 4; ++u) mx = fmaxf(mx, fmaxf(vh[u], vw[u]));
#pragma unroll
  for (int s = 16; s >= 1; s >>= 1) mx = fmaxf(mx, __shfl_xor(mx, s, 32));
  float sum = 0.f;
#pragma unroll
  for (int u = 0; u < 4; ++u) {
    vh[u] = __expf(vh[u] - mx);
    vw[u] = __expf(vw[u] - mx);
    sum += vh[u] + vw[u];
  }
#pragma unroll
  for (int s = 16; s >= 1; s >>= 1) sum += __shfl_xor(sum, s, 32);
  const float inv = 1.0f / sum;
  bf16_t* aH = (bf16_t*)rH;
  bf16_t* aW = (bf16_t*)rW;
#pragma unroll
  for (int u = 0; u < 4; ++u) {
    aH[lane * 4 + u] = (bf16_t)(vh[u] * inv);
    aW[lane * 4 + u] = (bf16_t)(vw[u] * inv);
  }
}

// =============== K4a: out = x + gamma * (V_col @ aH^T) ===============
// Per (b, w, c-tile of 128): Out[c][i] = sum_j v[b][c][j][w] * aH[i][j]
__global__ __launch_bounds__(256) void outH_kernel(
    const bf16_t* __restrict__ qkv, const float* __restrict__ eH,
    const float* __restrict__ x, const float* __restrict__ gamma,
    float* __restrict__ out)
{
  __shared__ __align__(16) bf16_t Vs[128][128 + 8];   // [c][j]  (A: [M][K])
  __shared__ __align__(16) bf16_t Ts[128][128 + 8];   // [i][j]  (B: [N][K])
  const int w  = blockIdx.x;
  const int c0 = blockIdx.y * 128;
  const int b  = blockIdx.z;
  const int t  = threadIdx.x;
  const bf16_t* vbase = qkv + ((size_t)b * MQKV_ + 128 + c0) * HW_;
  const bf16_t* aH = (const bf16_t*)(eH + ((size_t)b * 128 + w) * 128 * 128); // row stride 256

  // attention rows: straight 16B-chunk copies (async path if available)
  for (int idx = t; idx < 2048; idx += 256) {
    const int i = idx >> 4;
    const int c16 = (idx & 15) * 8;
    ldsCopy16(&Ts[i][c16], aH + (size_t)i * 256 + c16);
  }
  // V column gather (strided, L2-resident across blocks)
  for (int idx = t; idx < 128 * 128; idx += 256) {
    const int j = idx & 127;
    const int c = idx >> 7;
    Vs[c][j] = vbase[(size_t)c * HW_ + (size_t)j * W_ + w];
  }
  ldsCopyFence();
  __syncthreads();

  const int wave = t >> 5;
  const int wc = (wave >> 1) * 32;   // 4 waves along c
  const int wi = (wave & 1) * 64;    // 2 waves along i
  v8f acc[2][4] = {};
#pragma unroll
  for (int k0 = 0; k0 < 128; k0 += 32)
#pragma unroll
    for (int mi = 0; mi < 2; ++mi) {
      v16bf a = load_fragA(&Vs[0][0], wc + mi * 16, k0, 136);
#pragma unroll
      for (int ni = 0; ni < 4; ++ni) {
        v16bf bb = load_fragB(&Ts[0][0], k0, wi + ni * 16, 136);
        acc[mi][ni] = WMMA_BF16(a, bb, acc[mi][ni]);
      }
    }
  const float g = gamma[0];
  const int lane = t & 31;
  const int icol = lane & 15;
  const int moff = (lane & 16) ? 8 : 0;
#pragma unroll
  for (int mi = 0; mi < 2; ++mi)
#pragma unroll
    for (int ni = 0; ni < 4; ++ni)
#pragma unroll
      for (int r = 0; r < 8; ++r) {
        const int c = c0 + wc + mi * 16 + moff + r;
        const int i = wi + ni * 16 + icol;
        const size_t o = (((size_t)b * C_ + c) * H_ + i) * W_ + w;
        out[o] = x[o] + g * acc[mi][ni][r];
      }
}

// =============== K4b: out += gamma * (V_row @ aW^T) ===============
// Per (b, h, c-tile): Out[c][i] = sum_j v[b][c][h][j] * aW[i][j]
// V tile (128 rows x 256B, row stride HW_*2B) loaded by the Tensor Data Mover
// when available (wave 0 issues; TDM pads 16B per 256B -> [128][136] layout).
__global__ __launch_bounds__(256) void outW_kernel(
    const bf16_t* __restrict__ qkv, const float* __restrict__ eW,
    const float* __restrict__ gamma, float* __restrict__ out)
{
  __shared__ __align__(16) bf16_t Vs[128][128 + 8];   // [c][j]
  __shared__ __align__(16) bf16_t Ts[128][128 + 8];   // [i][j]
  const int h  = blockIdx.x;
  const int c0 = blockIdx.y * 128;
  const int b  = blockIdx.z;
  const int t  = threadIdx.x;
  const bf16_t* vrow = qkv + ((size_t)b * MQKV_ + 128 + c0) * HW_ + (size_t)h * W_;
  const bf16_t* aW = (const bf16_t*)(eW + ((size_t)b * 128 + h) * 128 * 128); // row stride 256

#if defined(USE_TDM)
  if ((t >> 5) == 0) {
    tdm_load_2d_bf16(&Vs[0][0], vrow, 128, 128, HW_);
    __builtin_amdgcn_s_wait_tensorcnt(0);
  }
  for (int idx = t; idx < 2048; idx += 256) {
    const int i = idx >> 4;
    const int c16 = (idx & 15) * 8;
    ldsCopy16(&Ts[i][c16], aW + (size_t)i * 256 + c16);
  }
#else
  for (int idx = t; idx < 2048; idx += 256) {
    const int i = idx >> 4;
    const int c16 = (idx & 15) * 8;
    ldsCopy16(&Ts[i][c16], aW + (size_t)i * 256 + c16);
    ldsCopy16(&Vs[i][c16], vrow + (size_t)i * HW_ + c16);
  }
#endif
  ldsCopyFence();
  __syncthreads();

  const int wave = t >> 5;
  const int wc = (wave >> 1) * 32;
  const int wi = (wave & 1) * 64;
  v8f acc[2][4] = {};
#pragma unroll
  for (int k0 = 0; k0 < 128; k0 += 32)
#pragma unroll
    for (int mi = 0; mi < 2; ++mi) {
      v16bf a = load_fragA(&Vs[0][0], wc + mi * 16, k0, 136);
#pragma unroll
      for (int ni = 0; ni < 4; ++ni) {
        v16bf bb = load_fragB(&Ts[0][0], k0, wi + ni * 16, 136);
        acc[mi][ni] = WMMA_BF16(a, bb, acc[mi][ni]);
      }
    }
  const float g = gamma[0];
  const int lane = t & 31;
  const int icol = lane & 15;
  const int moff = (lane & 16) ? 8 : 0;
#pragma unroll
  for (int mi = 0; mi < 2; ++mi)
#pragma unroll
    for (int ni = 0; ni < 4; ++ni)
#pragma unroll
      for (int r = 0; r < 8; ++r) {
        const int c = c0 + wc + mi * 16 + moff + r;
        const int i = wi + ni * 16 + icol;
        const size_t o = (((size_t)b * C_ + c) * H_ + h) * W_ + i;
        out[o] += g * acc[mi][ni][r];
      }
}

// =============== launch ===============
extern "C" void kernel_launch(void* const* d_in, const int* in_sizes, int n_in,
                              void* d_out, int out_size, void* d_ws, size_t ws_size,
                              hipStream_t stream) {
  (void)in_sizes; (void)n_in; (void)out_size; (void)ws_size;
  const float* x     = (const float*)d_in[0];
  const float* Wq    = (const float*)d_in[1];
  const float* Wk    = (const float*)d_in[2];
  const float* Wv    = (const float*)d_in[3];
  const float* gamma = (const float*)d_in[4];
  float* out = (float*)d_out;

  char* ws = (char*)d_ws;
  bf16_t* qkv = (bf16_t*)ws;                                 // 4*640*16384*2  = 83,886,080 B
  float*  eH  = (float*)(ws + (size_t)83886080);             // 33,554,432 B
  float*  eW  = (float*)(ws + (size_t)83886080 + 33554432);  // 33,554,432 B  (total ~151 MB)

  qkv_proj_kernel<<<dim3(HW_ / 128, MQKV_ / 64, B_), 256, 0, stream>>>(x, Wq, Wk, Wv, qkv);
  energy_kernel  <<<dim3(128, 2, B_),                256, 0, stream>>>(qkv, eH, eW);
  softmax_kernel <<<dim3(B_ * HW_ / 8),              256, 0, stream>>>(eH, eW);
  outH_kernel    <<<dim3(W_, C_ / 128, B_),          256, 0, stream>>>(qkv, eH, x, gamma, out);
  outW_kernel    <<<dim3(H_, C_ / 128, B_),          256, 0, stream>>>(qkv, eW, gamma, out);
}
